// SpectralMSA_30305289241151
// MI455X (gfx1250) — compile-verified, matrix operand order checked by reading, not run on hardware
//
#include <hip/hip_runtime.h>
#include <hip/hip_bf16.h>

// ---------------------------------------------------------------------------
// SpectralMSA (channel attention) for MI455X / gfx1250, wave32 + bf16 WMMA.
// All four contractions are NT GEMMs (K-contiguous operands) executed with
// v_wmma_f32_16x16x32_bf16, f32 accumulation; softmax in f32.
// Round 2: 64x64 wave tiles (4x4 WMMA accumulators) on the big GEMMs for 2x
// arithmetic intensity; logits kernel split 4-way over tokens (512 thr/blk).
// ---------------------------------------------------------------------------

typedef __bf16 bf16_t;
typedef __attribute__((ext_vector_type(16))) __bf16 v16bf;
typedef __attribute__((ext_vector_type(8)))  float  v8f;

#define WMMA_BF16(a, b, c) \
    __builtin_amdgcn_wmma_f32_16x16x32_bf16(false, (a), false, (b), (short)0, (c), false, false)

static constexpr int Bb = 4;
static constexpr int Nn = 4096;
static constexpr int Cc = 1024;
static constexpr int Hh = 16;
static constexpr int Dd = 64;

// ---- fp32 -> bf16 (round to nearest even) ----------------------------------
__device__ __forceinline__ unsigned short f2bf_bits(float f) {
    union { float f; unsigned u; } in; in.f = f;
    unsigned u = in.u;
    u += 0x7FFFu + ((u >> 16) & 1u);
    return (unsigned short)(u >> 16);
}
__device__ __forceinline__ bf16_t f2bf(float f) {
    union { unsigned short s; bf16_t b; } o; o.s = f2bf_bits(f); return o.b;
}

// ---- fragment load: row-major [row, ld] buffer, K contiguous ---------------
// A(16x32) layout: lane l -> row l%16, K chunks {(l/16)*8 .. +7} and {+16 .. +23}.
// Symmetric pattern used for B(32x16) fed from a row-major [n,k] buffer.
__device__ __forceinline__ v16bf load_frag_nt(const bf16_t* __restrict__ base,
                                              int ld, int row, int k0, int lane) {
    const int r  = row + (lane & 15);
    const int kg = k0 + ((lane >> 4) << 3);
    const bf16_t* p = base + (size_t)r * (size_t)ld + kg;
    union { v16bf v; struct { uint4 lo, hi; } q; } u;
    u.q.lo = *reinterpret_cast<const uint4*>(p);
    u.q.hi = *reinterpret_cast<const uint4*>(p + 16);
    return u.v;
}

// pack 8 accumulator floats (8 consecutive rows at one lane) into 8 bf16
__device__ __forceinline__ uint4 pack_bf16x8(const v8f& a) {
    union { uint4 q; unsigned short s[8]; } u;
#pragma unroll
    for (int j = 0; j < 8; ++j) u.s[j] = f2bf_bits(a[j]);
    return u.q;
}

// ---------------------------------------------------------------------------
// Kernel 0: fp32 -> bf16 cast, 8 elements per thread (b128 in/out)
// ---------------------------------------------------------------------------
__global__ __launch_bounds__(256) void cast_f32_bf16_v8(const float4* __restrict__ src,
                                                        uint4* __restrict__ dst, int n8) {
    int i = blockIdx.x * 256 + threadIdx.x;
    if (i < n8) {
        float4 lo = src[2 * i], hi = src[2 * i + 1];
        union { uint4 q; unsigned short s[8]; } u;
        u.s[0] = f2bf_bits(lo.x); u.s[1] = f2bf_bits(lo.y);
        u.s[2] = f2bf_bits(lo.z); u.s[3] = f2bf_bits(lo.w);
        u.s[4] = f2bf_bits(hi.x); u.s[5] = f2bf_bits(hi.y);
        u.s[6] = f2bf_bits(hi.z); u.s[7] = f2bf_bits(hi.w);
        dst[i] = u.q;
    }
}

// ---------------------------------------------------------------------------
// Kernel 1: QKV GEMM.  Y[m,o] = sum_c xb[m,c] * Wqkv[o,c]
//   m = b*4096+tok (16384), o = t*1024 + h*64 + e (3072), K = 1024.
// Scatter: Q,K stored transposed per head -> [bh, e/d, tok] (tok contiguous),
//          V stored [bh, tok, e] (e contiguous).
// Block: 128 thr = 4 waves (2x2), block tile 128x128, wave tile 64x64
// (4x4 WMMA accumulators -> 16 WMMA per 8 fragment loads).
// ---------------------------------------------------------------------------
__global__ __launch_bounds__(128, 1) void qkv_gemm(const bf16_t* __restrict__ xb,
                                                   const bf16_t* __restrict__ wb,
                                                   bf16_t* __restrict__ Qt,
                                                   bf16_t* __restrict__ Kt,
                                                   bf16_t* __restrict__ Vb) {
    const int lane = threadIdx.x & 31;
    const int wave = threadIdx.x >> 5;
    const int mBase = blockIdx.y * 128 + (wave >> 1) * 64;
    const int nBase = blockIdx.x * 128 + (wave & 1) * 64;

    v8f acc[4][4] = {};
    for (int k = 0; k < Cc; k += 32) {
        v16bf af[4], bfr[4];
#pragma unroll
        for (int i = 0; i < 4; ++i) af[i]  = load_frag_nt(xb, Cc, mBase + i * 16, k, lane);
#pragma unroll
        for (int i = 0; i < 4; ++i) bfr[i] = load_frag_nt(wb, Cc, nBase + i * 16, k, lane);
#pragma unroll
        for (int mi = 0; mi < 4; ++mi)
#pragma unroll
            for (int ni = 0; ni < 4; ++ni)
                acc[mi][ni] = WMMA_BF16(af[mi], bfr[ni], acc[mi][ni]);
    }

    // nBase is 64-aligned -> t, h uniform per wave; e spans 0..63
    const int t  = nBase >> 10;          // 0=Q 1=K 2=V
    const int h  = (nBase >> 6) & 15;
    const int b  = mBase >> 12;
    const int bh = b * Hh + h;
#pragma unroll
    for (int mi = 0; mi < 4; ++mi) {
#pragma unroll
        for (int ni = 0; ni < 4; ++ni) {
            const int m   = mBase + mi * 16 + ((lane >> 4) << 3); // 8 consecutive tokens
            const int tok = m & (Nn - 1);
            const int e   = ni * 16 + (lane & 15);
            if (t < 2) {
                bf16_t* dst = (t == 0 ? Qt : Kt);
                *reinterpret_cast<uint4*>(dst + ((size_t)(bh * Dd + e)) * Nn + tok) =
                    pack_bf16x8(acc[mi][ni]);
            } else {
                bf16_t* dst = Vb + ((size_t)bh * Nn + tok) * Dd + e;
#pragma unroll
                for (int j = 0; j < 8; ++j) dst[(size_t)j * Dd] = f2bf(acc[mi][ni][j]);
            }
        }
    }
}

// ---------------------------------------------------------------------------
// Kernel 2: attention logits + softmax per (b,h).
//   logits[d,e] = (sum_n Kt[d,n]*Qt[e,n]) / alpha[h];  softmax over e.
// One block per bh, 512 threads = 16 waves. Wave w: d-band (w&3)*16..+16,
// token slice (w>>2)*1024..+1024, accumulating into LDS plane (w>>2).
// Softmax pass sums the 4 planes. Result stored bf16 [bh, d, e].
// ---------------------------------------------------------------------------
__global__ __launch_bounds__(512) void attn_softmax(const bf16_t* __restrict__ Kt,
                                                    const bf16_t* __restrict__ Qt,
                                                    const float* __restrict__ alpha,
                                                    bf16_t* __restrict__ attnb) {
    __shared__ float logits[4][64][65];
    const int bh   = blockIdx.x;
    const int h    = bh & 15;
    const int lane = threadIdx.x & 31;
    const int wave = threadIdx.x >> 5;
    const bf16_t* Kb = Kt + (size_t)bh * Dd * Nn;
    const bf16_t* Qb = Qt + (size_t)bh * Dd * Nn;

    const int dBase = (wave & 3) * 16;
    const int slice = wave >> 2;          // token slice 0..3
    const int k0    = slice * (Nn / 4);
    v8f acc[4] = {};
    for (int k = k0; k < k0 + Nn / 4; k += 32) {
        v16bf a = load_frag_nt(Kb, Nn, dBase, k, lane);
#pragma unroll
        for (int ni = 0; ni < 4; ++ni) {
            v16bf b = load_frag_nt(Qb, Nn, ni * 16, k, lane);
            acc[ni] = WMMA_BF16(a, b, acc[ni]);
        }
    }

    const int d0 = dBase + ((lane >> 4) << 3);
#pragma unroll
    for (int ni = 0; ni < 4; ++ni) {
        const int e = ni * 16 + (lane & 15);
#pragma unroll
        for (int j = 0; j < 8; ++j) logits[slice][d0 + j][e] = acc[ni][j];
    }
    __syncthreads();

    if (threadIdx.x < 64) {
        const int d = threadIdx.x;
        const float inv_alpha = 1.0f / alpha[h];
        float row[64];
        float mx = -3.402823466e38f;
        for (int e = 0; e < 64; ++e) {
            float v = (logits[0][d][e] + logits[1][d][e] +
                       logits[2][d][e] + logits[3][d][e]) * inv_alpha;
            row[e] = v;
            mx = fmaxf(mx, v);
        }
        float s = 0.0f;
        for (int e = 0; e < 64; ++e) { float v = __expf(row[e] - mx); row[e] = v; s += v; }
        const float inv = 1.0f / s;
        bf16_t* dst = attnb + (size_t)bh * (Dd * Dd) + d * Dd;
        for (int e = 0; e < 64; ++e) dst[e] = f2bf(row[e] * inv);
    }
}

// ---------------------------------------------------------------------------
// Kernel 3: head output. O1[b,tok,h*64+d] = sum_e V[bh,tok,e] * attn[bh,d,e]
// Per (b,h): M=4096 tokens, N=64, K=64. Wave tile 64x64 (full N), block of
// 4 waves covers 256 tokens. Grid.x=bh, grid.y=token tile of 256.
// ---------------------------------------------------------------------------
__global__ __launch_bounds__(128, 1) void head_out(const bf16_t* __restrict__ Vb,
                                                   const bf16_t* __restrict__ attnb,
                                                   bf16_t* __restrict__ O1) {
    const int bh = blockIdx.x;
    const int b  = bh >> 4;
    const int h  = bh & 15;
    const bf16_t* V = Vb    + (size_t)bh * Nn * Dd;
    const bf16_t* A = attnb + (size_t)bh * (Dd * Dd);
    const int lane = threadIdx.x & 31;
    const int wave = threadIdx.x >> 5;
    const int mBase = blockIdx.y * 256 + wave * 64;

    v8f acc[4][4] = {};
#pragma unroll
    for (int k = 0; k < Dd; k += 32) {
        v16bf af[4], bfr[4];
#pragma unroll
        for (int i = 0; i < 4; ++i) af[i]  = load_frag_nt(V, Dd, mBase + i * 16, k, lane);
#pragma unroll
        for (int i = 0; i < 4; ++i) bfr[i] = load_frag_nt(A, Dd, i * 16, k, lane);
#pragma unroll
        for (int mi = 0; mi < 4; ++mi)
#pragma unroll
            for (int ni = 0; ni < 4; ++ni)
                acc[mi][ni] = WMMA_BF16(af[mi], bfr[ni], acc[mi][ni]);
    }
#pragma unroll
    for (int mi = 0; mi < 4; ++mi) {
#pragma unroll
        for (int ni = 0; ni < 4; ++ni) {
            const int tok = mBase + mi * 16 + ((lane >> 4) << 3);
            const int d   = ni * 16 + (lane & 15);
            bf16_t* dst = O1 + ((size_t)(b * Nn + tok)) * Cc + h * Dd + d;
#pragma unroll
            for (int j = 0; j < 8; ++j) dst[(size_t)j * Cc] = f2bf(acc[mi][ni][j]);
        }
    }
}

// ---------------------------------------------------------------------------
// Kernel 4: projection. out[m,o] = sum_c O1[m,c]*Wproj[o,c] + bproj[o], fp32.
// Block tile 128x128, wave tile 64x64.
// ---------------------------------------------------------------------------
__global__ __launch_bounds__(128, 1) void proj_gemm(const bf16_t* __restrict__ O1,
                                                    const bf16_t* __restrict__ wb,
                                                    const float* __restrict__ bias,
                                                    float* __restrict__ out) {
    const int lane = threadIdx.x & 31;
    const int wave = threadIdx.x >> 5;
    const int mBase = blockIdx.y * 128 + (wave >> 1) * 64;
    const int nBase = blockIdx.x * 128 + (wave & 1) * 64;

    v8f acc[4][4] = {};
    for (int k = 0; k < Cc; k += 32) {
        v16bf af[4], bfr[4];
#pragma unroll
        for (int i = 0; i < 4; ++i) af[i]  = load_frag_nt(O1, Cc, mBase + i * 16, k, lane);
#pragma unroll
        for (int i = 0; i < 4; ++i) bfr[i] = load_frag_nt(wb, Cc, nBase + i * 16, k, lane);
#pragma unroll
        for (int mi = 0; mi < 4; ++mi)
#pragma unroll
            for (int ni = 0; ni < 4; ++ni)
                acc[mi][ni] = WMMA_BF16(af[mi], bfr[ni], acc[mi][ni]);
    }
#pragma unroll
    for (int mi = 0; mi < 4; ++mi) {
#pragma unroll
        for (int ni = 0; ni < 4; ++ni) {
            const int o  = nBase + ni * 16 + (lane & 15);
            const int m  = mBase + mi * 16 + ((lane >> 4) << 3);
            const float bv = bias[o];
            float* dst = out + (size_t)m * Cc + o;
#pragma unroll
            for (int j = 0; j < 8; ++j) dst[(size_t)j * Cc] = acc[mi][ni][j] + bv;
        }
    }
}

// ---------------------------------------------------------------------------
// Launch
// ---------------------------------------------------------------------------
extern "C" void kernel_launch(void* const* d_in, const int* in_sizes, int n_in,
                              void* d_out, int out_size, void* d_ws, size_t ws_size,
                              hipStream_t stream) {
    const float* x     = (const float*)d_in[0];
    const float* Wqkv  = (const float*)d_in[3];
    const float* Wproj = (const float*)d_in[4];
    const float* bproj = (const float*)d_in[5];
    const float* alpha = (const float*)d_in[6];

    const size_t nX     = (size_t)Bb * Nn * Cc;      // 16,777,216
    const size_t nWqkv  = (size_t)3 * Cc * Cc;       //  3,145,728
    const size_t nWproj = (size_t)Cc * Cc;           //  1,048,576
    const size_t nHead  = (size_t)Bb * Hh * Nn * Dd; // 16,777,216 per Q/K/V
    const size_t nAttn  = (size_t)Bb * Hh * Dd * Dd;

    char* ws = (char*)d_ws;
    size_t off = 0;
    auto carve = [&](size_t elems) { bf16_t* p = (bf16_t*)(ws + off);
                                     off += ((elems * sizeof(bf16_t)) + 255) & ~(size_t)255;
                                     return p; };
    bf16_t* xb     = carve(nX);
    bf16_t* wqkvb  = carve(nWqkv);
    bf16_t* wprojb = carve(nWproj);
    bf16_t* Qt     = carve(nHead);   // [bh, e, tok]
    bf16_t* Kt     = carve(nHead);   // [bh, d, tok]
    bf16_t* Vb     = carve(nHead);   // [bh, tok, e]
    bf16_t* attnb  = carve(nAttn);   // [bh, d, e]
    bf16_t* O1     = carve(nX);      // [b, tok, c]

    cast_f32_bf16_v8<<<dim3((unsigned)(nX / 8 / 256)), 256, 0, stream>>>(
        (const float4*)x, (uint4*)xb, (int)(nX / 8));
    cast_f32_bf16_v8<<<dim3((unsigned)(nWqkv / 8 / 256)), 256, 0, stream>>>(
        (const float4*)Wqkv, (uint4*)wqkvb, (int)(nWqkv / 8));
    cast_f32_bf16_v8<<<dim3((unsigned)(nWproj / 8 / 256)), 256, 0, stream>>>(
        (const float4*)Wproj, (uint4*)wprojb, (int)(nWproj / 8));

    qkv_gemm<<<dim3(3 * Cc / 128, Bb * Nn / 128), 128, 0, stream>>>(xb, wqkvb, Qt, Kt, Vb);
    attn_softmax<<<dim3(Bb * Hh), 512, 0, stream>>>(Kt, Qt, alpha, attnb);
    head_out<<<dim3(Bb * Hh, Nn / 256), 128, 0, stream>>>(Vb, attnb, O1);
    proj_gemm<<<dim3(Cc / 128, Bb * Nn / 128), 128, 0, stream>>>(O1, wprojb, bproj, (float*)d_out);
}
